// RRNNCell_50818053046611
// MI455X (gfx1250) — matrix operand order repaired and checked by reading, not compiled, not run on hardware
//
#include <hip/hip_runtime.h>
#include <stdint.h>

// ---------------------------------------------------------------------------
// RRNN cell for MI455X (gfx1250, wave32, WMMA).
//   Phase 1: u = x @ W via v_wmma_f32_16x16x32_bf16 (128x128 block tile,
//            8 waves x 8 WMMA tiles), hardware bf16 converts, fused
//            bias+sigmoid epilogue -> 5 gate planes [16384][512] in d_ws.
//   Phase 2: per-channel sequential scan over L with coalesced plane reads.
// ---------------------------------------------------------------------------

typedef __attribute__((ext_vector_type(16))) __bf16 v16bf;
typedef __attribute__((ext_vector_type(8)))  float  v8f;

#define L_SEQ   1024
#define BATCH   16
#define NIN     1024
#define NOUT    512
#define KGATES  5
#define NCOL    (NOUT * KGATES)      // 2560
#define MROWS   (L_SEQ * BATCH)      // 16384
#define PLANE   ((size_t)MROWS * NOUT)

#define MT 128          // block M tile
#define NT 128          // block N tile
#define NTILES (NT / 16)
#define KT 32           // K step (matches 16x16x32 bf16 WMMA)
#define LDA 40          // padded LDS stride (ushorts) for A tile rows
#define LDB 40          // padded LDS stride (ushorts) for B tile columns

__device__ __forceinline__ unsigned short f2bf(float f) {
    // hardware fp32 -> bf16 convert (v_cvt path on gfx1250)
    __bf16 h = (__bf16)f;
    return __builtin_bit_cast(unsigned short, h);
}

__device__ __forceinline__ float sigmoidf_dev(float x) {
    return 1.0f / (1.0f + __expf(-x));
}

__global__ __launch_bounds__(256)
void rrnn_gemm_gates(const float* __restrict__ x,
                     const float* __restrict__ w,
                     const float* __restrict__ bias,
                     float* __restrict__ ws) {
    __shared__ unsigned short Als[MT * LDA];   // bf16 A tile, row-major [m][k]
    __shared__ unsigned short Bls[NT * LDB];   // bf16 B tile, col-major [n][k]

    const int tid  = threadIdx.x;
    const int lane = tid & 31;
    const int wave = tid >> 5;              // 0..7, each wave = 16 M-rows
    const int m0   = blockIdx.y * MT;
    const int n0   = blockIdx.x * NT;

    // A-tile loader: 256 threads cover 128 rows x 2 half-rows of 16 k
    const int ar = tid >> 1;                // 0..127
    const int ah = (tid & 1) * 16;          // k offset 0 or 16
    // B-tile loader: 256 threads cover 32 k-rows x 8 col-groups of 16
    const int bk = tid >> 3;                // 0..31
    const int bn = (tid & 7) * 16;          // 0..112

    v8f acc[NTILES];
    #pragma unroll
    for (int t = 0; t < NTILES; ++t)
        acc[t] = (v8f){0.f,0.f,0.f,0.f,0.f,0.f,0.f,0.f};

    const int mrow = wave * 16 + (lane & 15);
    const int half = lane >> 4;             // 0 or 1

    for (int k0 = 0; k0 < NIN; k0 += KT) {
        // ---- stage A tile: x[m0+ar][k0+ah .. +15] -> bf16 LDS ----
        const float* ax = x + (size_t)(m0 + ar) * NIN + k0 + ah;
        float4 a0 = ((const float4*)ax)[0];
        float4 a1 = ((const float4*)ax)[1];
        float4 a2 = ((const float4*)ax)[2];
        float4 a3 = ((const float4*)ax)[3];
        if (k0 + KT < NIN) __builtin_prefetch(ax + KT, 0, 1);  // global_prefetch_b8

        union { unsigned short us[16]; uint4 q[2]; } pk;
        pk.us[0]=f2bf(a0.x); pk.us[1]=f2bf(a0.y); pk.us[2]=f2bf(a0.z); pk.us[3]=f2bf(a0.w);
        pk.us[4]=f2bf(a1.x); pk.us[5]=f2bf(a1.y); pk.us[6]=f2bf(a1.z); pk.us[7]=f2bf(a1.w);
        pk.us[8]=f2bf(a2.x); pk.us[9]=f2bf(a2.y); pk.us[10]=f2bf(a2.z); pk.us[11]=f2bf(a2.w);
        pk.us[12]=f2bf(a3.x); pk.us[13]=f2bf(a3.y); pk.us[14]=f2bf(a3.z); pk.us[15]=f2bf(a3.w);
        uint4* adst = (uint4*)&Als[ar * LDA + ah];
        adst[0] = pk.q[0];
        adst[1] = pk.q[1];

        // ---- stage B tile: w[k0+bk][n0+bn .. +15] -> bf16 LDS (transposed) ----
        const float* bx = w + (size_t)(k0 + bk) * NCOL + n0 + bn;
        float4 b0 = ((const float4*)bx)[0];
        float4 b1 = ((const float4*)bx)[1];
        float4 b2 = ((const float4*)bx)[2];
        float4 b3 = ((const float4*)bx)[3];
        Bls[(bn +  0) * LDB + bk] = f2bf(b0.x);
        Bls[(bn +  1) * LDB + bk] = f2bf(b0.y);
        Bls[(bn +  2) * LDB + bk] = f2bf(b0.z);
        Bls[(bn +  3) * LDB + bk] = f2bf(b0.w);
        Bls[(bn +  4) * LDB + bk] = f2bf(b1.x);
        Bls[(bn +  5) * LDB + bk] = f2bf(b1.y);
        Bls[(bn +  6) * LDB + bk] = f2bf(b1.z);
        Bls[(bn +  7) * LDB + bk] = f2bf(b1.w);
        Bls[(bn +  8) * LDB + bk] = f2bf(b2.x);
        Bls[(bn +  9) * LDB + bk] = f2bf(b2.y);
        Bls[(bn + 10) * LDB + bk] = f2bf(b2.z);
        Bls[(bn + 11) * LDB + bk] = f2bf(b2.w);
        Bls[(bn + 12) * LDB + bk] = f2bf(b3.x);
        Bls[(bn + 13) * LDB + bk] = f2bf(b3.y);
        Bls[(bn + 14) * LDB + bk] = f2bf(b3.z);
        Bls[(bn + 15) * LDB + bk] = f2bf(b3.w);

        __syncthreads();

        // ---- A fragment per ISA layout: lanes 0-15 k=0..7,16..23 ; 16-31 +8 ----
        v16bf afrag;
        {
            union { v16bf v; uint4 q[2]; } u;
            const unsigned short* ap = &Als[mrow * LDA + half * 8];
            u.q[0] = *(const uint4*)ap;          // k = half*8 + 0..7
            u.q[1] = *(const uint4*)(ap + 16);   // k = 16 + half*8 + 0..7
            afrag = u.v;
        }

        // ---- 8 N-tiles of 16; B fragment k = half*8 + 0..7, 16 + half*8 + 0..7 (matches A).
        //      Load all fragments first so ds_load overlaps v_wmma issue. ----
        v16bf bfrag[NTILES];
        #pragma unroll
        for (int t = 0; t < NTILES; ++t) {
            union { v16bf v; uint4 q[2]; } u;
        const unsigned short* bp =
                &Bls[(t * 16 + (lane & 15)) * LDB + half * 8];
            u.q[0] = *(const uint4*)bp;
            u.q[1] = *(const uint4*)(bp + 16);
            bfrag[t] = u.v;
        }
        #pragma unroll
        for (int t = 0; t < NTILES; ++t) {
            acc[t] = __builtin_amdgcn_wmma_f32_16x16x32_bf16(
                false, afrag, false, bfrag[t], (short)0, acc[t], false, false);
        }

        __syncthreads();
    }

    // ---- epilogue: column -> (gate, channel); fuse sigmoid for gates 2,3,4 ----
    const int m_hi = (lane >> 4) * 8;   // D layout: lanes 16-31 hold M+8
    const int ncl  = lane & 15;
    #pragma unroll
    for (int t = 0; t < NTILES; ++t) {
        const int col  = n0 + t * 16 + ncl;
        const int g    = col % KGATES;
        const int chan = col / KGATES;
        const float bv = (g >= 2) ? bias[g * NOUT + chan] : 0.0f;
        float* plane = ws + (size_t)g * PLANE;
        #pragma unroll
        for (int p = 0; p < 8; ++p) {
            const int m = m0 + wave * 16 + m_hi + p;
            float v = acc[t][p];
            if (g >= 2) v = sigmoidf_dev(v + bv);
            plane[(size_t)m * NOUT + chan] = v;
        }
    }
}

__global__ __launch_bounds__(256)
void rrnn_scan(const float* __restrict__ ws,
               const float* __restrict__ bias_eps,
               const float* __restrict__ bias_final,
               float* __restrict__ out) {
    const int ch = blockIdx.x * 256 + threadIdx.x;   // 0..8191 = b*512+n
    const int n  = ch & (NOUT - 1);

    const float eps  = sigmoidf_dev(bias_eps[n]);
    const float rho0 = sigmoidf_dev(bias_final[2 * n + 0]) * 2.0f;
    const float rho1 = sigmoidf_dev(bias_final[2 * n + 1]) * 2.0f;

    const float* pu1 = ws + 0 * PLANE;
    const float* pu2 = ws + 1 * PLANE;
    const float* pf1 = ws + 2 * PLANE;
    const float* pf2 = ws + 3 * PLANE;
    const float* pog = ws + 4 * PLANE;

    float c1 = 0.0f, c2 = 0.0f;
    for (int l = 0; l < L_SEQ; ++l) {
        const size_t idx = (size_t)l * (BATCH * NOUT) + ch;  // row m = l*B+b
        const float u1 = pu1[idx];
        const float u2 = pu2[idx];
        const float f1 = pf1[idx];
        const float f2 = pf2[idx];
        const float og = pog[idx];
        const float c1n = c1 * f1 + u1;
        const float c2n = c2 * f2 + (eps + c1) * u2;
        c1 = c1n;
        c2 = c2n;
        const float cs = c1 * rho0 + c2 * rho1;
        out[idx] = tanhf(og * cs);
    }
    out[PLANE + ch]                = c1;   // c1_final [B][NOUT]
    out[PLANE + BATCH * NOUT + ch] = c2;   // c2_final [B][NOUT]
}

extern "C" void kernel_launch(void* const* d_in, const int* in_sizes, int n_in,
                              void* d_out, int out_size, void* d_ws, size_t ws_size,
                              hipStream_t stream) {
    const float* x    = (const float*)d_in[0];
    const float* w    = (const float*)d_in[1];
    const float* bias = (const float*)d_in[2];
    const float* beps = (const float*)d_in[3];
    const float* bfin = (const float*)d_in[4];
    float* out = (float*)d_out;
    float* ws  = (float*)d_ws;   // 5 planes * 16384*512 floats = 167.8 MB

    dim3 grid_gemm(NCOL / NT, MROWS / MT);   // (20, 128)
    rrnn_gemm_gates<<<grid_gemm, 256, 0, stream>>>(x, w, bias, ws);

    rrnn_scan<<<(BATCH * NOUT) / 256, 256, 0, stream>>>(ws, beps, bfin, out);
}
